// CrossAttnBlock_13898514170138
// MI455X (gfx1250) — matrix operand order repaired and checked by reading, not compiled, not run on hardware
//
#include <hip/hip_runtime.h>

// ---------------------------------------------------------------------------
// CrossAttnBlock for MI455X (gfx1250): bf16 WMMA everywhere.
//   B=2, C=256, D=16, W=32, H=32 -> N=16384 spatial, SN=256 tokens, CS=256,
//   HEADS=8, DH=32.
// ---------------------------------------------------------------------------

typedef unsigned short u16;
typedef __attribute__((ext_vector_type(16))) __bf16       v16bf;
typedef __attribute__((ext_vector_type(8)))  float        v8f;
typedef __attribute__((ext_vector_type(4)))  unsigned int v4u;
typedef __attribute__((ext_vector_type(4)))  float        v4f;

union FragBF {           // one WMMA A/B operand: 16 bf16 = 8 VGPRs
  v4u   u[2];
  v16bf v;
  u16   e[16];
};

union Pack8 {            // 8 bf16 = 16 bytes
  v4u v;
  u16 e[8];
};

__device__ __forceinline__ u16 f2bf(float f) {   // round-to-nearest-even
  union { float f; unsigned int u; } a; a.f = f;
  unsigned int u = a.u;
  u += 0x7FFFu + ((u >> 16) & 1u);
  return (u16)(u >> 16);
}

#define CC    256
#define NN    16384     // D*W*H
#define SNT   256
#define NHEAD 8
#define DH    32

// ---------------------------------------------------------------------------
// Kernel 1: channel-first LayerNorm of x -> bf16 hn (B, C, N)
// ---------------------------------------------------------------------------
__global__ __launch_bounds__(256) void ln_x_kernel(
    const float* __restrict__ x, const float* __restrict__ lw,
    const float* __restrict__ lb, u16* __restrict__ hn) {
  int gidx = blockIdx.x * 256 + threadIdx.x;       // 0 .. B*N-1
  int b = gidx >> 14;
  int n = gidx & (NN - 1);
  const float* xb = x + (size_t)b * CC * NN + n;
  float s = 0.f, ss = 0.f;
  for (int c = 0; c < CC; ++c) {
    float v = xb[(size_t)c * NN];
    s += v; ss += v * v;
  }
  float u   = s * (1.f / CC);
  float var = ss * (1.f / CC) - u * u;
  float inv = rsqrtf(var + 1e-6f);
  u16* hb = hn + (size_t)b * CC * NN + n;
  for (int c = 0; c < CC; ++c) {
    float v = (xb[(size_t)c * NN] - u) * inv * lw[c] + lb[c];
    hb[(size_t)c * NN] = f2bf(v);
  }
}

// ---------------------------------------------------------------------------
// Kernel 2: token LayerNorm of s -> bf16 snT (B, CS, SN)   (transposed)
// ---------------------------------------------------------------------------
__global__ __launch_bounds__(256) void ln_s_kernel(
    const float* __restrict__ s, const float* __restrict__ lw,
    const float* __restrict__ lb, u16* __restrict__ snT) {
  __shared__ float red[256];
  int b = blockIdx.x >> 8;
  int t = blockIdx.x & 255;
  int c = threadIdx.x;
  float v = s[((size_t)(b * SNT + t)) * CC + c];
  red[c] = v; __syncthreads();
  for (int off = 128; off; off >>= 1) {
    if (c < off) red[c] += red[c + off];
    __syncthreads();
  }
  float u = red[0] * (1.f / CC);
  __syncthreads();
  float d = v - u;
  red[c] = d * d; __syncthreads();
  for (int off = 128; off; off >>= 1) {
    if (c < off) red[c] += red[c + off];
    __syncthreads();
  }
  float inv = rsqrtf(red[0] * (1.f / CC) + 1e-6f);
  snT[((size_t)(b * CC + c)) * SNT + t] = f2bf(d * inv * lw[c] + lb[c]);
}

// ---------------------------------------------------------------------------
// Kernel 3: generic 256-K GEMM, D = Wt(256xo,256c) x B(256c, Ntot), bf16 WMMA.
//   Block: 256 threads (8 waves), tile 64(M) x 128(N), K stepped by 32.
//   mode 0: bf16 out[o*Ntot + n]                       (V projection)
//   mode 1: bf16 out[((o/32)*Ntot + n)*32 + o%32]      (Q / K, head-packed)
//   mode 2: f32  out = acc + bias[o] + resid           (proj_out + residual)
// ---------------------------------------------------------------------------
__global__ __launch_bounds__(256) void gemm256_bf16(
    const float* __restrict__ Wt,    // (256,256) row-major (o,c)
    const u16*  __restrict__ Bsrc,   // (batch,256,Ntot) bf16
    int Ntot,
    u16*   __restrict__ outb,
    float* __restrict__ outf,
    const float* __restrict__ bias,
    const float* __restrict__ resid,
    int mode, float scale) {
  __shared__ u16 A_lds[64 * 40];     // 64 rows(o) x 32 k   (+8 pad)
  __shared__ u16 BT_lds[128 * 40];   // 128 rows(n) x 32 k  (+8 pad), transposed

  const int tid   = threadIdx.x;
  const int lane  = tid & 31;
  const int wv    = tid >> 5;
  const int msub  = wv & 3;          // which 16-row M subtile
  const int npair = wv >> 2;         // which group of 4 N subtiles
  const int g     = lane >> 4;
  const int qn    = lane & 15;

  const int n0 = blockIdx.x * 128;
  const int m0 = blockIdx.y * 64;
  const size_t bofs = (size_t)blockIdx.z * CC * (size_t)Ntot;
  const u16* Bb = Bsrc + bofs;

  v8f acc[4] = {{}, {}, {}, {}};

  for (int k0 = 0; k0 < 256; k0 += 32) {
    __syncthreads();
    { // stage A: fp32 weights -> bf16 LDS, one 16B ds store per thread
      int o  = tid >> 2;
      int cc = (tid & 3) * 8;
      const float* src = Wt + (size_t)(m0 + o) * 256 + k0 + cc;
      v4f f0 = *(const v4f*)(src);
      v4f f1 = *(const v4f*)(src + 4);
      if (k0 + 32 < 256) __builtin_prefetch(src + 32, 0, 0);
      Pack8 pk;
      pk.e[0] = f2bf(f0.x); pk.e[1] = f2bf(f0.y);
      pk.e[2] = f2bf(f0.z); pk.e[3] = f2bf(f0.w);
      pk.e[4] = f2bf(f1.x); pk.e[5] = f2bf(f1.y);
      pk.e[6] = f2bf(f1.z); pk.e[7] = f2bf(f1.w);
      *(v4u*)&A_lds[o * 40 + cc] = pk.v;
    }
    { // stage B transposed: two k-rows per thread, packed b32 LDS stores
      int cr = (tid >> 4) * 2;       // even k row: 0,2,...,30
      int nn = (tid & 15) * 8;       // 0..120
      const u16* src0 = Bb + (size_t)(k0 + cr) * Ntot + n0 + nn;
      const u16* src1 = src0 + Ntot;
      Pack8 r0, r1;
      r0.v = *(const v4u*)(src0);
      r1.v = *(const v4u*)(src1);
      if (k0 + 32 < 256) {
        __builtin_prefetch(src0 + (size_t)32 * Ntot, 0, 0);
        __builtin_prefetch(src1 + (size_t)32 * Ntot, 0, 0);
      }
      unsigned int* bt32 = (unsigned int*)&BT_lds[0];
      #pragma unroll
      for (int e = 0; e < 8; ++e) {
        unsigned int pkv =
            (unsigned int)r0.e[e] | ((unsigned int)r1.e[e] << 16);
        bt32[(nn + e) * 20 + (cr >> 1)] = pkv;   // ((nn+e)*40 + cr)/2
      }
    }
    __syncthreads();

    FragBF af;   // A: lane row m = qn, K run g*16..g*16+15
    const u16* arow = &A_lds[(msub * 16 + qn) * 40];
    af.u[0] = *(const v4u*)(arow + g * 16);
    af.u[1] = *(const v4u*)(arow + g * 16 + 8);
    #pragma unroll
    for (int i = 0; i < 4; ++i) {
      int nsub = npair * 4 + i;
      const u16* brow = &BT_lds[(nsub * 16 + qn) * 40 + g * 16];
      FragBF bf;   // B: lane col n = qn, K run g*16..g*16+15
      bf.u[0] = *(const v4u*)(brow);
      bf.u[1] = *(const v4u*)(brow + 8);
      acc[i] = __builtin_amdgcn_wmma_f32_16x16x32_bf16(
          false, af.v, false, bf.v, (short)0, acc[i], false, false);
    }
  }

  // D layout: M = r + 8*g, N = qn
  #pragma unroll
  for (int i = 0; i < 4; ++i) {
    int nsub = npair * 4 + i;
    #pragma unroll
    for (int r = 0; r < 8; ++r) {
      int o = m0 + msub * 16 + r + 8 * g;
      int n = n0 + nsub * 16 + qn;
      float val = acc[i][r] * scale;
      if (mode == 0) {
        outb[bofs + (size_t)o * Ntot + n] = f2bf(val);
      } else if (mode == 1) {
        outb[bofs + ((size_t)(o >> 5) * Ntot + n) * 32 + (o & 31)] = f2bf(val);
      } else {
        size_t idx = bofs + (size_t)o * Ntot + n;
        outf[idx] = val + bias[o] + resid[idx];
      }
    }
  }
}

// ---------------------------------------------------------------------------
// Kernel 4: attention.  One wave handles a 16-row tile of N for one (b,head).
//   scores(16 x 256) = q(16 x 32) x k(32 x 256)   -> 16 WMMAs (K=32)
//   softmax over 256 in registers (shfl_xor within 16-lane half-groups)
//   out(16 x 32)     = attn(16 x 256) x vT(256 x 32) -> 16 WMMAs (8 K-steps)
// ---------------------------------------------------------------------------
__global__ __launch_bounds__(32) void attn_kernel(
    const u16* __restrict__ qq,   // (B,H,N,32)  bf16, pre-scaled by dh^-0.5
    const u16* __restrict__ kk,   // (B,H,SN,32) bf16
    const u16* __restrict__ vv,   // (B,C,SN)    bf16  (C = H*32)
    u16* __restrict__ ao) {       // (B,C,N)     bf16
  __shared__ u16 attn_lds[16 * 264];

  const int lane = threadIdx.x;
  const int g    = lane >> 4;
  const int qn   = lane & 15;
  const int n0   = blockIdx.x * 16;
  const int h    = blockIdx.y;
  const int b    = blockIdx.z;
  const int bh   = b * NHEAD + h;

  // Q A-fragment: lane row n = n0+qn, k run g*16..g*16+15
  FragBF qf;
  const u16* qrow = qq + ((size_t)bh * NN + n0 + qn) * 32;
  qf.u[0] = *(const v4u*)(qrow + g * 16);
  qf.u[1] = *(const v4u*)(qrow + g * 16 + 8);

  v8f sc[16];
  #pragma unroll
  for (int j = 0; j < 16; ++j) {
    FragBF kf;  // B: lane col t = j*16+qn, k run d = g*16..g*16+15
    const u16* krow = kk + ((size_t)bh * SNT + j * 16 + qn) * 32 + g * 16;
    kf.u[0] = *(const v4u*)(krow);
    kf.u[1] = *(const v4u*)(krow + 8);
    v8f z = {};
    sc[j] = __builtin_amdgcn_wmma_f32_16x16x32_bf16(
        false, qf.v, false, kf.v, (short)0, z, false, false);
  }

  // softmax: element (j, r) belongs to row m = r+8*g, col j*16+qn
  float mx[8], si[8];
  #pragma unroll
  for (int r = 0; r < 8; ++r) {
    float m = sc[0][r];
    #pragma unroll
    for (int j = 1; j < 16; ++j) m = fmaxf(m, sc[j][r]);
    #pragma unroll
    for (int d = 1; d < 16; d <<= 1) m = fmaxf(m, __shfl_xor(m, d, 32));
    mx[r] = m;
  }
  #pragma unroll
  for (int r = 0; r < 8; ++r) {
    float s = 0.f;
    #pragma unroll
    for (int j = 0; j < 16; ++j) {
      float e = __expf(sc[j][r] - mx[r]);   // cache exp in-place
      sc[j][r] = e;
      s += e;
    }
    #pragma unroll
    for (int d = 1; d < 16; d <<= 1) s += __shfl_xor(s, d, 32);
    si[r] = 1.f / s;
  }
  #pragma unroll
  for (int j = 0; j < 16; ++j) {
    #pragma unroll
    for (int r = 0; r < 8; ++r) {
      attn_lds[(r + 8 * g) * 264 + j * 16 + qn] = f2bf(sc[j][r] * si[r]);
    }
  }
  __syncthreads();

  v8f oacc0 = {}, oacc1 = {};
  #pragma unroll
  for (int c = 0; c < 8; ++c) {
    int s0 = c * 32;
    FragBF af;   // attn A-fragment for K(=s) window s0..s0+31
    const u16* arow = &attn_lds[qn * 264 + s0];
    af.u[0] = *(const v4u*)(arow + g * 16);
    af.u[1] = *(const v4u*)(arow + g * 16 + 8);
    #pragma unroll
    for (int t = 0; t < 2; ++t) {
      FragBF vf;  // B[s][d]: lane col d = t*16+qn, k run s = s0+g*16..
      const u16* vrow =
          vv + ((size_t)(bh * 32 + t * 16 + qn)) * SNT + s0 + g * 16;
      vf.u[0] = *(const v4u*)(vrow);
      vf.u[1] = *(const v4u*)(vrow + 8);
      if (t == 0)
        oacc0 = __builtin_amdgcn_wmma_f32_16x16x32_bf16(
            false, af.v, false, vf.v, (short)0, oacc0, false, false);
      else
        oacc1 = __builtin_amdgcn_wmma_f32_16x16x32_bf16(
            false, af.v, false, vf.v, (short)0, oacc1, false, false);
    }
  }

  // store: D rows = n-local r+8*g, cols = d-local qn; channel = h*32 + d
  #pragma unroll
  for (int t = 0; t < 2; ++t) {
    #pragma unroll
    for (int r = 0; r < 8; ++r) {
      float val = (t == 0) ? oacc0[r] : oacc1[r];
      int n   = n0 + r + 8 * g;
      int cch = h * 32 + t * 16 + qn;
      ao[((size_t)(b * CC + cch)) * NN + n] = f2bf(val);
    }
  }
}

// ---------------------------------------------------------------------------
// Host-side launcher
// ---------------------------------------------------------------------------
extern "C" void kernel_launch(void* const* d_in, const int* in_sizes, int n_in,
                              void* d_out, int out_size, void* d_ws,
                              size_t ws_size, hipStream_t stream) {
  const float* x     = (const float*)d_in[0];
  const float* s     = (const float*)d_in[1];
  const float* ln_w  = (const float*)d_in[2];
  const float* ln_b  = (const float*)d_in[3];
  const float* lns_w = (const float*)d_in[4];
  const float* lns_b = (const float*)d_in[5];
  const float* Wq    = (const float*)d_in[6];
  const float* Wk    = (const float*)d_in[7];
  const float* Wv    = (const float*)d_in[8];
  const float* Wo    = (const float*)d_in[9];
  const float* bo    = (const float*)d_in[10];
  float* out = (float*)d_out;

  // workspace carve (bf16 tensors)
  char* w = (char*)d_ws;
  u16* hn  = (u16*)w; w += (size_t)2 * CC * NN  * 2;   // 16 MB
  u16* snT = (u16*)w; w += (size_t)2 * CC * SNT * 2;   // 256 KB
  u16* qq  = (u16*)w; w += (size_t)2 * CC * NN  * 2;   // 16 MB
  u16* kk  = (u16*)w; w += (size_t)2 * CC * SNT * 2;   // 256 KB
  u16* vv  = (u16*)w; w += (size_t)2 * CC * SNT * 2;   // 256 KB
  u16* ao  = (u16*)w; w += (size_t)2 * CC * NN  * 2;   // 16 MB

  const float qscale = 0.17677669529663687f;           // 1/sqrt(32)

  ln_x_kernel<<<(2 * NN) / 256, 256, 0, stream>>>(x, ln_w, ln_b, hn);
  ln_s_kernel<<<2 * SNT, 256, 0, stream>>>(s, lns_w, lns_b, snT);

  // K: mode 1 (head-packed (b,h,t,32));  V: mode 0 ((b,c,s))
  gemm256_bf16<<<dim3(SNT / 128, 4, 2), 256, 0, stream>>>(
      Wk, snT, SNT, kk, nullptr, nullptr, nullptr, 1, 1.0f);
  gemm256_bf16<<<dim3(SNT / 128, 4, 2), 256, 0, stream>>>(
      Wv, snT, SNT, vv, nullptr, nullptr, nullptr, 0, 1.0f);
  // Q: mode 1, scale folded in
  gemm256_bf16<<<dim3(NN / 128, 4, 2), 256, 0, stream>>>(
      Wq, hn, NN, qq, nullptr, nullptr, nullptr, 1, qscale);

  attn_kernel<<<dim3(NN / 16, NHEAD, 2), 32, 0, stream>>>(qq, kk, vv, ao);

  // proj_out + bias + residual -> f32 d_out
  gemm256_bf16<<<dim3(NN / 128, 4, 2), 256, 0, stream>>>(
      Wo, ao, NN, nullptr, out, bo, x, 2, 1.0f);
}
